// PerformerAttention_79860621902509
// MI455X (gfx1250) — matrix operand order repaired and checked by reading, not compiled, not run on hardware
//
#include <hip/hip_runtime.h>
#include <hip/hip_bf16.h>

#define BATCH  4
#define SEQ    4096
#define DMODEL 1024
#define MFEAT  256

typedef __attribute__((ext_vector_type(16))) __bf16        bf16x16;
typedef __attribute__((ext_vector_type(8)))  float         f32x8;
typedef __attribute__((ext_vector_type(8)))  unsigned short us8;
typedef __attribute__((ext_vector_type(4)))  unsigned int  uint4v;

__device__ __forceinline__ unsigned short f2bf_bits(float f) {
    return __builtin_bit_cast(unsigned short, (__bf16)f);   // v_cvt to bf16, RNE
}
__device__ __forceinline__ unsigned pack_bf16(float lo, float hi) {
    unsigned short a = f2bf_bits(lo), b = f2bf_bits(hi);
    return (unsigned)a | ((unsigned)b << 16);
}
__device__ __forceinline__ float bf2f(unsigned short h) {
    return __uint_as_float(((unsigned)h) << 16);
}

union FragU { bf16x16 v; us8 h[2]; };

// A-matrix 16x32 bf16 fragment (ISA layout): lanes 0-15 rows M, K=kb..kb+7 then kb+16..kb+23
__device__ __forceinline__ bf16x16 load_frag_A(const unsigned short* lds, int stride, int rowBase) {
    int lane = threadIdx.x & 31;
    const unsigned short* p = lds + (rowBase + (lane & 15)) * stride + ((lane < 16) ? 0 : 8);
    FragU u;
    u.h[0] = *(const us8*)(p);
    u.h[1] = *(const us8*)(p + 16);
    return u.v;
}
// B-matrix 32x16 bf16 fragment: lane -> column N, K=kb..kb+7 then kb+16..kb+23 (+8 for lanes 16-31) (lds stored [col][k])
__device__ __forceinline__ bf16x16 load_frag_B(const unsigned short* lds, int stride, int colBase) {
    int lane = threadIdx.x & 31;
    const unsigned short* p = lds + (colBase + (lane & 15)) * stride + ((lane < 16) ? 0 : 8);
    FragU u;
    u.h[0] = *(const us8*)(p);
    u.h[1] = *(const us8*)(p + 16);
    return u.v;
}

#define WMMA_BF16(A, B, C) \
    __builtin_amdgcn_wmma_f32_16x16x32_bf16(false, (A), false, (B), (short)0, (C), false, false)

// --- CDNA5 async global->LDS 16-byte copy (per-lane), tracked by ASYNCcnt ---
__device__ __forceinline__ void async_copy16(unsigned short* lds_dst, const unsigned short* gsrc) {
#if defined(__gfx1250__)
    unsigned laddr = (unsigned)(uintptr_t)lds_dst;            // flat low bits == LDS offset
    unsigned long long gaddr = (unsigned long long)(uintptr_t)gsrc;
    asm volatile("global_load_async_to_lds_b128 %0, %1, off" ::"v"(laddr), "v"(gaddr) : "memory");
#else
    *(us8*)lds_dst = *(const us8*)gsrc;
#endif
}
__device__ __forceinline__ void wait_async() {
#if defined(__gfx1250__)
    asm volatile("s_wait_asynccnt 0" ::: "memory");
#endif
}

// ---------------------------------------------------------------------------
// Kernel 1/2: rf = exp(preScale * X @ P^T) * postScale ; bf16 out, optionally
// transposed ([m][l]).  Block tile 128(l) x 64(m), K = DMODEL in steps of 32.
// ---------------------------------------------------------------------------
template <int TRANSPOSE_OUT>
__global__ __launch_bounds__(256) void rf_kernel(const float* __restrict__ X,
                                                 const float* __restrict__ P,
                                                 unsigned short* __restrict__ out,
                                                 float preScale, float postScale) {
    constexpr int STR  = 40;   // LDS row stride (ushorts) = 80B, 16B aligned
    constexpr int TSTR = 136;  // transpose buffer stride  = 272B, 16B aligned
    __shared__ __align__(16) unsigned short smem[8704];
    unsigned short* As = smem;             // 128 x STR
    unsigned short* Ps = smem + 128 * STR; // 64 x STR

    const int b  = blockIdx.z;
    const int l0 = blockIdx.y * 128;
    const int m0 = blockIdx.x * 64;
    const int t  = threadIdx.x;
    const int wid = t >> 5, lane = t & 31;
    const int wr = wid >> 1, wc = wid & 1;
    const int hi = lane >> 4, colN = lane & 15;

    f32x8 acc[2][2] = {};

    for (int kk = 0; kk < DMODEL; kk += 32) {
        {   // stage X tile 128x32, scaled f32 -> packed bf16, 2x b128 stores
            int row = t >> 1, half = t & 1;
            const float* src = X + ((size_t)b * SEQ + (l0 + row)) * DMODEL + kk + half * 16;
            unsigned u[8];
#pragma unroll
            for (int j = 0; j < 4; ++j) {
                float4 v = *(const float4*)(src + j * 4);
                u[j * 2 + 0] = pack_bf16(v.x * preScale, v.y * preScale);
                u[j * 2 + 1] = pack_bf16(v.z * preScale, v.w * preScale);
            }
            unsigned int* dst = (unsigned int*)(As + row * STR + half * 16);
            *(uint4v*)(dst)     = uint4v{u[0], u[1], u[2], u[3]};
            *(uint4v*)(dst + 4) = uint4v{u[4], u[5], u[6], u[7]};
        }
        {   // stage P tile 64x32 ([m][k]), 1x b128 store
            int m = t >> 2, part = t & 3;
            const float* src = P + (size_t)(m0 + m) * DMODEL + kk + part * 8;
            float4 v0 = *(const float4*)(src);
            float4 v1 = *(const float4*)(src + 4);
            uint4v u = {pack_bf16(v0.x, v0.y), pack_bf16(v0.z, v0.w),
                        pack_bf16(v1.x, v1.y), pack_bf16(v1.z, v1.w)};
            *(uint4v*)(Ps + m * STR + part * 8) = u;
        }
        __syncthreads();
        bf16x16 a0 = load_frag_A(As, STR, wr * 32);
        bf16x16 a1 = load_frag_A(As, STR, wr * 32 + 16);
        bf16x16 b0 = load_frag_B(Ps, STR, wc * 32);
        bf16x16 b1 = load_frag_B(Ps, STR, wc * 32 + 16);
        acc[0][0] = WMMA_BF16(a0, b0, acc[0][0]);
        acc[0][1] = WMMA_BF16(a0, b1, acc[0][1]);
        acc[1][0] = WMMA_BF16(a1, b0, acc[1][0]);
        acc[1][1] = WMMA_BF16(a1, b1, acc[1][1]);
        __syncthreads();
    }

    if (!TRANSPOSE_OUT) {
#pragma unroll
        for (int i = 0; i < 2; ++i)
#pragma unroll
            for (int j = 0; j < 2; ++j)
#pragma unroll
                for (int r = 0; r < 8; ++r) {
                    int l = l0 + wr * 32 + i * 16 + r + hi * 8;
                    int m = m0 + wc * 32 + j * 16 + colN;
                    out[((size_t)b * SEQ + l) * MFEAT + m] =
                        f2bf_bits(__expf(acc[i][j][r]) * postScale);
                }
    } else {
        // transpose through LDS (pack pairs along l), then coalesced [m][l] stores
#pragma unroll
        for (int i = 0; i < 2; ++i)
#pragma unroll
            for (int j = 0; j < 2; ++j) {
                int mLoc  = wc * 32 + j * 16 + colN;
                int lBase = wr * 32 + i * 16 + hi * 8;
#pragma unroll
                for (int r = 0; r < 8; r += 2) {
                    unsigned u = pack_bf16(__expf(acc[i][j][r + 0]) * postScale,
                                           __expf(acc[i][j][r + 1]) * postScale);
                    *(unsigned*)(smem + mLoc * TSTR + lBase + r) = u;
                }
            }
        __syncthreads();
        int m = t >> 2, seg = t & 3;
        const unsigned short* srcRow = smem + m * TSTR + seg * 32;
        unsigned short* dstRow = out + ((size_t)b * MFEAT + m0 + m) * SEQ + l0 + seg * 32;
#pragma unroll
        for (int j = 0; j < 32; j += 8)
            *(us8*)(dstRow + j) = *(const us8*)(srcRow + j);
    }
}

// ---------------------------------------------------------------------------
// Kernel 3: KV^T[d][m] = (rf_k^T @ V)^T in bf16.
// A = rfkT [m][l] bf16 (async copy to LDS), B = V [l][d] f32 staged transposed.
// Block tile 128(m) x 64(d), K = SEQ in steps of 32.
// ---------------------------------------------------------------------------
__global__ __launch_bounds__(256) void kv_kernel(const unsigned short* __restrict__ rfkT,
                                                 const float* __restrict__ V,
                                                 unsigned short* __restrict__ kvT) {
    constexpr int STR  = 40;
    constexpr int TSTR = 136;
    __shared__ __align__(16) unsigned short smem[8704];
    unsigned short* As = smem;
    unsigned short* Bs = smem + 128 * STR;

    const int b  = blockIdx.z;
    const int m0 = blockIdx.y * 128;
    const int d0 = blockIdx.x * 64;
    const int t  = threadIdx.x;
    const int wid = t >> 5, lane = t & 31;
    const int wr = wid >> 1, wc = wid & 1;
    const int hi = lane >> 4, colN = lane & 15;

    f32x8 acc[2][2] = {};

    for (int ll = 0; ll < SEQ; ll += 32) {
        {   // A tile: rfkT[m0+row][ll..] -> LDS via async b128 copies
            int row = t >> 1, half = t & 1;
            const unsigned short* src =
                rfkT + ((size_t)b * MFEAT + m0 + row) * SEQ + ll + half * 16;
            unsigned short* dst = As + row * STR + half * 16;
            async_copy16(dst, src);
            async_copy16(dst + 8, src + 8);
        }
        {   // B tile: V[l][d] -> Bs[d][l], pack l-pairs -> b32 stores
            int l = (t >> 4) * 2;      // 0,2,..,30
            int dg = (t & 15) * 4;     // 0,4,..,60
            const float* s0 = V + ((size_t)b * SEQ + ll + l) * DMODEL + d0 + dg;
            float4 r0 = *(const float4*)(s0);
            float4 r1 = *(const float4*)(s0 + DMODEL);
#pragma unroll
            for (int j = 0; j < 4; ++j) {
                float a = (&r0.x)[j], c = (&r1.x)[j];
                *(unsigned*)(Bs + (dg + j) * STR + l) = pack_bf16(a, c);
            }
        }
        wait_async();
        __syncthreads();
        bf16x16 a0 = load_frag_A(As, STR, wr * 32);
        bf16x16 a1 = load_frag_A(As, STR, wr * 32 + 16);
        bf16x16 b0 = load_frag_B(Bs, STR, wc * 32);
        bf16x16 b1 = load_frag_B(Bs, STR, wc * 32 + 16);
        acc[0][0] = WMMA_BF16(a0, b0, acc[0][0]);
        acc[0][1] = WMMA_BF16(a0, b1, acc[0][1]);
        acc[1][0] = WMMA_BF16(a1, b0, acc[1][0]);
        acc[1][1] = WMMA_BF16(a1, b1, acc[1][1]);
        __syncthreads();
    }

    // KV[m][d] tile -> kvT[d][m] via LDS transpose (pack pairs along m), bf16
#pragma unroll
    for (int i = 0; i < 2; ++i)
#pragma unroll
        for (int j = 0; j < 2; ++j) {
            int dLoc  = wc * 32 + j * 16 + colN;
            int mBase = wr * 32 + i * 16 + hi * 8;
#pragma unroll
            for (int r = 0; r < 8; r += 2) {
                unsigned u = pack_bf16(acc[i][j][r + 0], acc[i][j][r + 1]);
                *(unsigned*)(smem + dLoc * TSTR + mBase + r) = u;
            }
        }
    __syncthreads();
    int d = t >> 2, seg = t & 3;
    const unsigned short* srcRow = smem + d * TSTR + seg * 32;
    unsigned short* dstRow = kvT + ((size_t)b * DMODEL + d0 + d) * MFEAT + m0 + seg * 32;
#pragma unroll
    for (int j = 0; j < 32; j += 8)
        *(us8*)(dstRow + j) = *(const us8*)(srcRow + j);
}

// ---------------------------------------------------------------------------
// Kernel 4: Z[b][m] = sum_l rfkT[b][m][l]   (wave per row, b128 loads)
// ---------------------------------------------------------------------------
__global__ __launch_bounds__(256) void z_kernel(const unsigned short* __restrict__ rfkT,
                                                float* __restrict__ Z) {
    int wid = threadIdx.x >> 5, lane = threadIdx.x & 31;
    int row = blockIdx.x * 8 + wid;  // 0 .. BATCH*MFEAT-1
    const unsigned short* src = rfkT + (size_t)row * SEQ;
    float s = 0.f;
#pragma unroll 4
    for (int j = 0; j < SEQ / 256; ++j) {
        us8 v = *(const us8*)(src + (j * 32 + lane) * 8);
#pragma unroll
        for (int e = 0; e < 8; ++e) s += bf2f(v[e]);
    }
#pragma unroll
    for (int off = 16; off; off >>= 1) s += __shfl_xor(s, off, 32);
    if (lane == 0) Z[row] = s;
}

// ---------------------------------------------------------------------------
// Kernel 5: rnorm[b][l] = 1 / (dot(rf_q[b][l][:], Z[b][:]) + EPS)
// ---------------------------------------------------------------------------
__global__ __launch_bounds__(256) void norm_kernel(const unsigned short* __restrict__ rfq,
                                                   const float* __restrict__ Z,
                                                   float* __restrict__ rnorm) {
    int wid = threadIdx.x >> 5, lane = threadIdx.x & 31;
    int row = blockIdx.x * 8 + wid;  // 0 .. BATCH*SEQ-1
    int b = row >> 12;               // SEQ = 4096
    const unsigned short* src = rfq + (size_t)row * MFEAT;
    const float* zb = Z + b * MFEAT;
    us8 v = *(const us8*)(src + lane * 8);
    float s = 0.f;
#pragma unroll
    for (int e = 0; e < 8; ++e) s += bf2f(v[e]) * zb[lane * 8 + e];
#pragma unroll
    for (int off = 16; off; off >>= 1) s += __shfl_xor(s, off, 32);
    if (lane == 0) rnorm[row] = __builtin_amdgcn_rcpf(s + 1e-6f);
}

// ---------------------------------------------------------------------------
// Kernel 6: out[l][d] = (rf_q @ KV)[l][d] * rnorm[l]   (fp32 output)
// A = rfq [l][m] bf16, B from kvT[d][m] (row-major in k=m). K = MFEAT = 256.
// ---------------------------------------------------------------------------
__global__ __launch_bounds__(256) void out_kernel(const unsigned short* __restrict__ rfq,
                                                  const unsigned short* __restrict__ kvT,
                                                  const float* __restrict__ rnorm,
                                                  float* __restrict__ out) {
    constexpr int STR = 40;
    __shared__ __align__(16) unsigned short smem[8704];
    unsigned short* As = smem;
    unsigned short* Bs = smem + 128 * STR;

    const int b  = blockIdx.z;
    const int l0 = blockIdx.y * 128;
    const int d0 = blockIdx.x * 64;
    const int t  = threadIdx.x;
    const int wid = t >> 5, lane = t & 31;
    const int wr = wid >> 1, wc = wid & 1;
    const int hi = lane >> 4, colN = lane & 15;

    f32x8 acc[2][2] = {};

    for (int mm = 0; mm < MFEAT; mm += 32) {
        {   // A tile: rfq[l0+row][mm..] (async b128 copies)
            int row = t >> 1, half = t & 1;
            const unsigned short* src =
                rfq + ((size_t)b * SEQ + l0 + row) * MFEAT + mm + half * 16;
            unsigned short* dst = As + row * STR + half * 16;
            async_copy16(dst, src);
            async_copy16(dst + 8, src + 8);
        }
        {   // B tile: kvT[d0+d][mm..] -> Bs[d][k=m] (async b128 copy)
            int d = t >> 2, part = t & 3;
            const unsigned short* src =
                kvT + ((size_t)b * DMODEL + d0 + d) * MFEAT + mm + part * 8;
            async_copy16(Bs + d * STR + part * 8, src);
        }
        wait_async();
        __syncthreads();
        bf16x16 a0 = load_frag_A(As, STR, wr * 32);
        bf16x16 a1 = load_frag_A(As, STR, wr * 32 + 16);
        bf16x16 b0 = load_frag_B(Bs, STR, wc * 32);
        bf16x16 b1 = load_frag_B(Bs, STR, wc * 32 + 16);
        acc[0][0] = WMMA_BF16(a0, b0, acc[0][0]);
        acc[0][1] = WMMA_BF16(a0, b1, acc[0][1]);
        acc[1][0] = WMMA_BF16(a1, b0, acc[1][0]);
        acc[1][1] = WMMA_BF16(a1, b1, acc[1][1]);
        __syncthreads();
    }

#pragma unroll
    for (int i = 0; i < 2; ++i)
#pragma unroll
        for (int r = 0; r < 8; ++r) {
            int l = l0 + wr * 32 + i * 16 + r + hi * 8;
            float rc = rnorm[(size_t)b * SEQ + l];
#pragma unroll
            for (int j = 0; j < 2; ++j) {
                int d = d0 + wc * 32 + j * 16 + colN;
                out[((size_t)b * SEQ + l) * DMODEL + d] = acc[i][j][r] * rc;
            }
        }
}

// ---------------------------------------------------------------------------
extern "C" void kernel_launch(void* const* d_in, const int* in_sizes, int n_in,
                              void* d_out, int out_size, void* d_ws, size_t ws_size,
                              hipStream_t stream) {
    const float* Q = (const float*)d_in[0];
    const float* K = (const float*)d_in[1];
    const float* V = (const float*)d_in[2];
    const float* P = (const float*)d_in[3];
    float* out = (float*)d_out;

    char* ws = (char*)d_ws;
    size_t off = 0;
    auto walloc = [&](size_t bytes) {
        void* p = ws + off;
        off = (off + bytes + 255) & ~(size_t)255;
        return p;
    };
    unsigned short* rfq   = (unsigned short*)walloc((size_t)BATCH * SEQ * MFEAT * 2);
    unsigned short* rfkT  = (unsigned short*)walloc((size_t)BATCH * SEQ * MFEAT * 2);
    unsigned short* kvT   = (unsigned short*)walloc((size_t)BATCH * DMODEL * MFEAT * 2);
    float* Zbuf           = (float*)walloc((size_t)BATCH * MFEAT * 4);
    float* rnormBuf       = (float*)walloc((size_t)BATCH * SEQ * 4);

    const float preScale = 1.0f / 32.0f;  // 1/sqrt(DMODEL)
    const float postQ    = 1.0f / 16.0f;  // 1/sqrt(MFEAT)

    dim3 blk(256);
    rf_kernel<0><<<dim3(MFEAT / 64, SEQ / 128, BATCH), blk, 0, stream>>>(Q, P, rfq, preScale, postQ);
    rf_kernel<1><<<dim3(MFEAT / 64, SEQ / 128, BATCH), blk, 0, stream>>>(K, P, rfkT, preScale, 1.0f);
    z_kernel<<<dim3(BATCH * MFEAT / 8), blk, 0, stream>>>(rfkT, Zbuf);
    kv_kernel<<<dim3(DMODEL / 64, MFEAT / 128, BATCH), blk, 0, stream>>>(rfkT, V, kvT);
    norm_kernel<<<dim3(BATCH * SEQ / 8), blk, 0, stream>>>(rfq, Zbuf, rnormBuf);
    out_kernel<<<dim3(DMODEL / 64, SEQ / 128, BATCH), blk, 0, stream>>>(rfq, kvT, rnormBuf, out);
}
